// VectorQuantizer_34359738651
// MI455X (gfx1250) — matrix operand order repaired and checked by reading, not compiled, not run on hardware
//
#include <hip/hip_runtime.h>
#include <stdint.h>

typedef __bf16 v16bf __attribute__((ext_vector_type(16)));
typedef float  v8f   __attribute__((ext_vector_type(8)));

#define ROWS   16384
#define DDIM   512
#define KCODES 8192

// workspace layout (bytes)
#define WS_XB    0u                      // 16384*512*2  z_e as bf16
#define WS_WB    16777216u               // 8192*512*2   W as bf16
#define WS_WN    25165824u               // 8192*4       0.5*||w||^2
#define WS_PACK  25198592u               // 16384*8      packed (score,~idx)
#define WS_SUM   25329664u               // 4            sum of squared error

// LDS tile geometry: 128 codes x 64 d-values bf16, 144B (36 dwords) per code
#define LDS_STRIDE_DW 36
#define LDS_BUF_DW    (128 * LDS_STRIDE_DW)   // 4608 dwords = 18KB per buffer

__device__ __forceinline__ unsigned short f2bf(float f) {
  unsigned int u = __float_as_uint(f);
  unsigned int r = ((u >> 16) & 1u) + 0x7FFFu;   // round-to-nearest-even
  return (unsigned short)((u + r) >> 16);
}

struct Frag32 { uint4 lo, hi; };   // 32 bytes -> one v16bf operand

// ---------------------------------------------------------------- init
__global__ void vq_init(unsigned long long* __restrict__ packed,
                        float* __restrict__ sum) {
  int g = blockIdx.x * 256 + threadIdx.x;
  if (g < ROWS) packed[g] = 0ull;
  if (g == 0) *sum = 0.0f;
}

// ---------------------------------------------------------------- W -> bf16 + 0.5*||w||^2
__global__ __launch_bounds__(256) void vq_prep_w(
    const float* __restrict__ W, unsigned int* __restrict__ Wb32,
    float* __restrict__ wn) {
  __shared__ float red[256];
  int row = blockIdx.x, t = threadIdx.x;
  float f0 = W[row * DDIM + 2 * t];
  float f1 = W[row * DDIM + 2 * t + 1];
  Wb32[row * 256 + t] = (unsigned)f2bf(f0) | ((unsigned)f2bf(f1) << 16);
  red[t] = f0 * f0 + f1 * f1;
  __syncthreads();
  for (int st = 128; st > 0; st >>= 1) {
    if (t < st) red[t] += red[t + st];
    __syncthreads();
  }
  if (t == 0) wn[row] = 0.5f * red[0];
}

// ---------------------------------------------------------------- z_e -> bf16
__global__ __launch_bounds__(256) void vq_prep_x(
    const float* __restrict__ Z, unsigned int* __restrict__ Xb32) {
  int g = blockIdx.x * 256 + threadIdx.x;   // one packed dword each
  float f0 = Z[2 * g], f1 = Z[2 * g + 1];
  Xb32[g] = (unsigned)f2bf(f0) | ((unsigned)f2bf(f1) << 16);
}

// ---------------------------------------------------------------- main: GEMM + argmax via WMMA
// grid: (ROWS/128, KCODES/1024); 256 threads = 8 waves; wave handles 16 rows.
// W tiles staged with global_load_async_to_lds_b128 into double-buffered LDS:
// stage of block kb+1 overlaps the 16 WMMAs of block kb.
__global__ __launch_bounds__(256) void vq_argmin_wmma(
    const uint4* __restrict__ Xb4, const uint4* __restrict__ Wb4,
    const float* __restrict__ wn, unsigned long long* __restrict__ packed) {
  __shared__ unsigned int wlds[2 * LDS_BUF_DW];   // 36 KB total

  const int tid  = threadIdx.x;
  const int lane = tid & 31;
  const int wave = tid >> 5;
  const int half = lane >> 4;       // 0: lanes 0-15, 1: lanes 16-31
  const int ln   = lane & 15;
  const int arow = blockIdx.x * 128 + wave * 16 + ln;   // A-matrix row (M = ln)
  const int aRowBase = arow * 64;   // uint4 per row = 512*2/16

  // async stage: 128 codes x 8 uint4 (64 d-values) -> LDS buffer `buf`
  auto stage = [&](int buf, int cb, int kb) {
#pragma unroll
    for (int i = 0; i < 4; ++i) {
      int j = tid + 256 * i;            // 1024 uint4 total
      int code = j >> 3, q = j & 7;
      unsigned ldsOff = (unsigned)(size_t)(
          &wlds[buf * LDS_BUF_DW + code * LDS_STRIDE_DW + q * 4]);
      unsigned long long ga = (unsigned long long)(size_t)(
          &Wb4[(size_t)(cb + code) * 64 + kb * 8 + q]);
      asm volatile("global_load_async_to_lds_b128 %0, %1, off"
                   :: "v"(ldsOff), "v"(ga) : "memory");
    }
  };

  float bestv[8];
  int   besti[8];
#pragma unroll
  for (int r = 0; r < 8; ++r) { bestv[r] = -3.402823466e38f; besti[r] = 0; }

  for (int c2 = 0; c2 < 8; ++c2) {
    const int codebase = blockIdx.y * 1024 + c2 * 128;
    v8f acc[8];
    const v8f zero8 = {0.f, 0.f, 0.f, 0.f, 0.f, 0.f, 0.f, 0.f};
#pragma unroll
    for (int nt = 0; nt < 8; ++nt) acc[nt] = zero8;

    stage(0, codebase, 0);              // prologue: first d-block of this chunk

    for (int kb = 0; kb < 8; ++kb) {    // 8 d-blocks of 64 (full D=512)
      const int cur = kb & 1;
      // my stage loads for the current buffer have retired (in-order), then
      // barrier => every wave's stage done AND every wave's previous compute done
      asm volatile("s_wait_asynccnt 0" ::: "memory");
      __syncthreads();
      if (kb < 7) stage(cur ^ 1, codebase, kb + 1);   // overlaps with compute below
      if (kb == 0) {                    // warm L2 for next chunk's codes
        const char* pf = reinterpret_cast<const char*>(
            &Wb4[(size_t)(codebase + 128 + (tid & 127)) * 64]);
        __builtin_prefetch(pf, 0, 1);
      }

      // Preload both A fragments for this d-block (one global clause)
      // A 16x32 bf16 layout: lanes 0-15 hold K 0-7 & 16-23; lanes 16-31 K 8-15 & 24-31
      Frag32 af[2];
#pragma unroll
      for (int ks = 0; ks < 2; ++ks) {
        af[ks].lo = Xb4[aRowBase + kb * 8 + ks * 4 + half];
        af[ks].hi = Xb4[aRowBase + kb * 8 + ks * 4 + half + 2];
      }

      const int bufDw = cur * LDS_BUF_DW;
#pragma unroll
      for (int ks = 0; ks < 2; ++ks) {  // k-steps of 32 within the block
        v16bf a = __builtin_bit_cast(v16bf, af[ks]);
        // all 8 B fragments in distinct registers -> ds_load clause + partial waits
        Frag32 bfr[8];
#pragma unroll
        for (int nt = 0; nt < 8; ++nt) {
          int o = bufDw + (nt * 16 + ln) * LDS_STRIDE_DW + ks * 16 + half * 4;
          bfr[nt].lo = *reinterpret_cast<const uint4*>(&wlds[o]);
          bfr[nt].hi = *reinterpret_cast<const uint4*>(&wlds[o + 8]);
        }
#pragma unroll
        for (int nt = 0; nt < 8; ++nt) {
          v16bf b = __builtin_bit_cast(v16bf, bfr[nt]);
          acc[nt] = __builtin_amdgcn_wmma_f32_16x16x32_bf16(
              false, a, false, b, (short)0, acc[nt], false, false);
        }
      }
    }

    // score = x.w - 0.5||w||^2 ; running per-row argmax
#pragma unroll
    for (int nt = 0; nt < 8; ++nt) {
      int code = codebase + nt * 16 + ln;
      float wnh = wn[code];
#pragma unroll
      for (int r = 0; r < 8; ++r) {
        float v = acc[nt][r] - wnh;
        if (v > bestv[r]) { bestv[r] = v; besti[r] = code; }  // strict > keeps lowest idx
      }
    }
  }

  // reduce across the 16 lanes holding the same row; C row = r + 8*half
#pragma unroll
  for (int r = 0; r < 8; ++r) {
    float v = bestv[r];
    int   i = besti[r];
#pragma unroll
    for (int m = 8; m >= 1; m >>= 1) {
      float ov = __shfl_xor(v, m, 32);
      int   oi = __shfl_xor(i, m, 32);
      if (ov > v || (ov == v && oi < i)) { v = ov; i = oi; }
    }
    if (ln == 0) {
      int outRow = blockIdx.x * 128 + wave * 16 + r + 8 * half;
      unsigned int u = __float_as_uint(v);
      u = (u & 0x80000000u) ? ~u : (u | 0x80000000u);   // monotonic float->uint
      unsigned long long p =
          ((unsigned long long)u << 32) | (unsigned long long)(~(unsigned int)i);
      atomicMax(&packed[outRow], p);   // ties: larger ~idx == smaller idx wins
    }
  }
}

// ---------------------------------------------------------------- gather z_q + squared-error sum
__global__ __launch_bounds__(128) void vq_gather_loss(
    const unsigned long long* __restrict__ packed,
    const float* __restrict__ W, const float* __restrict__ ze,
    float* __restrict__ out, float* __restrict__ sum) {
  __shared__ float red[128];
  int row = blockIdx.x;
  int t = threadIdx.x;
  unsigned long long p = packed[row];
  int idx = (int)(~(unsigned int)(p & 0xFFFFFFFFull)) & (KCODES - 1);

  const float4* wr = reinterpret_cast<const float4*>(W + (size_t)idx * DDIM);
  const float4* xr = reinterpret_cast<const float4*>(ze + (size_t)row * DDIM);
  float4* orow = reinterpret_cast<float4*>(out + (size_t)row * DDIM);

  float4 w4 = wr[t];
  float4 x4 = xr[t];
  orow[t] = w4;                               // z_q_st forward value == z_q
  float dx = w4.x - x4.x, dy = w4.y - x4.y, dz = w4.z - x4.z, dw = w4.w - x4.w;
  red[t] = dx * dx + dy * dy + dz * dz + dw * dw;
  __syncthreads();
  for (int st = 64; st > 0; st >>= 1) {
    if (t < st) red[t] += red[t + st];
    __syncthreads();
  }
  if (t == 0) {
    atomicAdd(sum, red[0]);
    out[(size_t)ROWS * DDIM + row] = (float)idx;   // ids as float
  }
}

// ---------------------------------------------------------------- scalars
__global__ void vq_finalize(const float* __restrict__ sum, float* __restrict__ out) {
  float mse = *sum * (1.0f / 8388608.0f);          // mean over B*N*D
  size_t o = (size_t)ROWS * DDIM + ROWS;
  out[o + 0] = 1.25f * mse;  // vq_loss = codebook + 0.25*commit
  out[o + 1] = mse;          // codebook_loss
  out[o + 2] = mse;          // commit_loss
  out[o + 3] = 0.0f;         // orth_loss
  out[o + 4] = 1.25f * mse;  // loss_total
}

// ---------------------------------------------------------------- launch
extern "C" void kernel_launch(void* const* d_in, const int* in_sizes, int n_in,
                              void* d_out, int out_size, void* d_ws, size_t ws_size,
                              hipStream_t stream) {
  (void)in_sizes; (void)n_in; (void)out_size; (void)ws_size;
  const float* z_e = (const float*)d_in[0];
  const float* W   = (const float*)d_in[1];
  float* out = (float*)d_out;
  char* ws = (char*)d_ws;

  unsigned int* Xb32 = (unsigned int*)(ws + WS_XB);
  unsigned int* Wb32 = (unsigned int*)(ws + WS_WB);
  float* wn = (float*)(ws + WS_WN);
  unsigned long long* packed = (unsigned long long*)(ws + WS_PACK);
  float* sum = (float*)(ws + WS_SUM);

  vq_init<<<64, 256, 0, stream>>>(packed, sum);
  vq_prep_w<<<KCODES, 256, 0, stream>>>(W, Wb32, wn);
  vq_prep_x<<<ROWS, 256, 0, stream>>>(z_e, Xb32);

  dim3 grid(ROWS / 128, KCODES / 1024);
  vq_argmin_wmma<<<grid, 256, 0, stream>>>(
      (const uint4*)Xb32, (const uint4*)Wb32, wn, packed);

  vq_gather_loss<<<ROWS, 128, 0, stream>>>(packed, W, z_e, out, sum);
  vq_finalize<<<1, 1, 0, stream>>>(sum, out);
}
